// GraphSAGE_14929306321143
// MI455X (gfx1250) — matrix operand order repaired and checked by reading, not compiled, not run on hardware
//
#include <hip/hip_runtime.h>

typedef __attribute__((ext_vector_type(16))) _Float16 v16h;
typedef __attribute__((ext_vector_type(8)))  float    v8f;

#define DIN 128   // feature width of both GEMM inputs (self and mean halves)

// ---------------------------------------------------------------- zero
__global__ void sage_zero(float* __restrict__ p, int n) {
  int i = blockIdx.x * blockDim.x + threadIdx.x;
  int stride = gridDim.x * blockDim.x;
  for (; i < n; i += stride) p[i] = 0.0f;
}

// ------------------------------------------------------- scatter-add
// One wave32 per edge: 32 lanes x 4 floats = 128 features.
__global__ void sage_scatter(const float* __restrict__ feat,
                             const int* __restrict__ src,
                             const int* __restrict__ dst,
                             float* __restrict__ agg,
                             float* __restrict__ cnt,
                             int E, int addCount) {
  int gid  = blockIdx.x * blockDim.x + threadIdx.x;
  int edge = gid >> 5;
  int lane = gid & 31;
  if (edge >= E) return;
  int s = src[edge];
  int d = dst[edge];
  const float4 v = *(const float4*)(feat + (size_t)s * DIN + lane * 4);
  float* to = agg + (size_t)d * DIN + lane * 4;
  atomicAdd(to + 0, v.x);
  atomicAdd(to + 1, v.y);
  atomicAdd(to + 2, v.z);
  atomicAdd(to + 3, v.w);
  if (addCount && lane == 0) atomicAdd(cnt + d, 1.0f);
}

// ---------------------------------------------------------- mean div
__global__ void sage_mean(float* __restrict__ agg, const float* __restrict__ cnt, int n) {
  int i = blockIdx.x * blockDim.x + threadIdx.x;
  if (i >= n) return;
  float c = cnt[i / DIN];
  agg[i] = agg[i] / fmaxf(c, 1.0f);
}

// ------------------------------------------------------------- GEMM
// out[N x NOUT] = [self | mean] (N x 256, f16) @ [Wa ; Wb] (256 x NOUT, f16) + ba + mask*bb
// 8 waves/block, each wave: 16 rows x NOUT cols, K=256 in 8 chunks of 32.
template <int NOUT, bool RELU>
__global__ __launch_bounds__(256) void sage_gemm(
    const float* __restrict__ selfF,   // N x 128
    const float* __restrict__ meanF,   // N x 128
    const float* __restrict__ Wa,      // 128 x NOUT (row-major, [k][n])
    const float* __restrict__ Wb,      // 128 x NOUT
    const float* __restrict__ ba,      // NOUT
    const float* __restrict__ bb,      // NOUT
    const float* __restrict__ cnt,     // N (in-degree)
    float* __restrict__ out,           // N x NOUT
    int N) {
  constexpr int KT = 256;
  constexpr int NTILES = NOUT / 16;
  __shared__ _Float16 Wt[NOUT * KT];   // W^T : [n][k], contiguous in k

  // Cooperative load + fp32->fp16 convert of the fused weight matrix.
  for (int idx = threadIdx.x; idx < NOUT * KT; idx += 256) {
    int n = idx >> 8;        // / KT
    int k = idx & (KT - 1);
    float w = (k < DIN) ? Wa[k * NOUT + n] : Wb[(k - DIN) * NOUT + n];
    Wt[n * KT + k] = (_Float16)w;
  }
  __syncthreads();

  const int wave = threadIdx.x >> 5;
  const int lane = threadIdx.x & 31;
  const int rowBase = blockIdx.x * 128 + wave * 16;
  const int mrow = lane & 15;
  const int row = rowBase + mrow;
  const bool rowValid = row < N;
  const int hiHalf = (lane < 16) ? 0 : 1;

  v8f acc[NTILES] = {};

  const unsigned* WtU = (const unsigned*)Wt;
  const int aKoff = hiHalf ? 16 : 0;   // A layout: lanes 0-15 K 0..15, lanes 16-31 K 16..31
  const int bKoff = hiHalf ? 16 : 0;   // B layout: lanes 0-15 K 0..15, lanes 16-31 K 16..31

#pragma unroll
  for (int kt = 0; kt < 8; ++kt) {
    const int kb = kt * 32;
    union { v16h v; _Float16 h[16]; } A;
    if (rowValid) {
      const float* srcRow = (kb < DIN) ? (selfF + (size_t)row * DIN + kb)
                                       : (meanF + (size_t)row * DIN + (kb - DIN));
      float af[16];
      *(float4*)(af + 0)  = ((const float4*)(srcRow + aKoff))[0];
      *(float4*)(af + 4)  = ((const float4*)(srcRow + aKoff))[1];
      *(float4*)(af + 8)  = ((const float4*)(srcRow + aKoff))[2];
      *(float4*)(af + 12) = ((const float4*)(srcRow + aKoff))[3];
#pragma unroll
      for (int j = 0; j < 16; ++j) A.h[j] = (_Float16)af[j];
    } else {
#pragma unroll
      for (int j = 0; j < 16; ++j) A.h[j] = (_Float16)0.0f;
    }

#pragma unroll
    for (int t = 0; t < NTILES; ++t) {
      const int n = t * 16 + mrow;
      union { v16h v; unsigned u[8]; } B;
      const unsigned* bp = WtU + ((n * KT + kb + bKoff) >> 1);
#pragma unroll
      for (int j = 0; j < 8; ++j) B.u[j] = bp[j];
      acc[t] = __builtin_amdgcn_wmma_f32_16x16x32_f16(
          false, A.v, false, B.v, (short)0, acc[t], false, false);
    }
  }

  // D layout: VGPR j -> row rowBase + j (+8 for lanes 16-31), col = t*16 + (lane&15).
#pragma unroll
  for (int t = 0; t < NTILES; ++t) {
    const int col = t * 16 + mrow;
    const float biasA = ba[col];
    const float biasB = bb[col];
#pragma unroll
    for (int j = 0; j < 8; ++j) {
      const int r = rowBase + j + (hiHalf ? 8 : 0);
      if (r < N) {
        float v = acc[t][j] + biasA + ((cnt[r] > 0.0f) ? biasB : 0.0f);
        if (RELU) v = fmaxf(v, 0.0f);
        out[(size_t)r * NOUT + col] = v;
      }
    }
  }
}

// ------------------------------------------------------------ driver
extern "C" void kernel_launch(void* const* d_in, const int* in_sizes, int n_in,
                              void* d_out, int out_size, void* d_ws, size_t ws_size,
                              hipStream_t stream) {
  (void)n_in; (void)out_size; (void)ws_size;
  const float* x  = (const float*)d_in[0];
  const int*   ei = (const int*)d_in[1];
  const float* W1 = (const float*)d_in[2];
  const float* b1 = (const float*)d_in[3];
  const float* W2 = (const float*)d_in[4];
  const float* b2 = (const float*)d_in[5];
  const float* W3 = (const float*)d_in[6];
  const float* b3 = (const float*)d_in[7];
  const float* W4 = (const float*)d_in[8];
  const float* b4 = (const float*)d_in[9];

  const int N = in_sizes[0] / DIN;       // 100000
  const int E = in_sizes[1] / 2;         // 600000
  const int* src = ei;
  const int* dst = ei + E;

  // Workspace layout: agg (N*128 f32) | cnt (N f32) | h (N*128 f32)
  float* agg = (float*)d_ws;
  float* cnt = agg + (size_t)N * DIN;
  float* h   = cnt + N;

  const int scatterBlocks = (E * 32 + 255) / 256;
  const int meanBlocks    = (N * DIN + 255) / 256;
  const int gemmBlocks    = (N + 127) / 128;

  // ---- Layer 1 ----
  sage_zero<<<2048, 256, 0, stream>>>(agg, N * DIN + N);  // agg + cnt contiguous
  sage_scatter<<<scatterBlocks, 256, 0, stream>>>(x, src, dst, agg, cnt, E, 1);
  sage_mean<<<meanBlocks, 256, 0, stream>>>(agg, cnt, N * DIN);
  sage_gemm<128, true><<<gemmBlocks, 256, 0, stream>>>(x, agg, W1, W2, b1, b2, cnt, h, N);

  // ---- Layer 2 ----
  sage_zero<<<2048, 256, 0, stream>>>(agg, N * DIN);
  sage_scatter<<<scatterBlocks, 256, 0, stream>>>(h, src, dst, agg, cnt, E, 0);
  sage_mean<<<meanBlocks, 256, 0, stream>>>(agg, cnt, N * DIN);
  sage_gemm<64, false><<<gemmBlocks, 256, 0, stream>>>(h, agg, W3, W4, b3, b4, cnt,
                                                       (float*)d_out, N);
}
